// MultiHeadAttention_16965120819454
// MI455X (gfx1250) — matrix operand order repaired and checked by reading, not compiled, not run on hardware
//
#include <hip/hip_runtime.h>

typedef __attribute__((ext_vector_type(16))) _Float16 v16h;
typedef __attribute__((ext_vector_type(8)))  _Float16 v8h;
typedef __attribute__((ext_vector_type(8)))  float    v8f;

#define NB   4
#define NS   2048
#define ND   256
#define NH   8
#define NDK  32
#define NROWS (NB * NS)   // 8192
#define NBH   (NB * NH)   // 32

__device__ __forceinline__ v8f wmma_f16(v16h a, v16h b, v8f c) {
  // D = A(16x32 f16) * B(32x16 f16) + C(16x16 f32)
  return __builtin_amdgcn_wmma_f32_16x16x32_f16(
      /*neg_a=*/false, a, /*neg_b=*/false, b,
      /*c_mod=*/(short)0, c, /*reuse_a=*/false, /*reuse_b=*/false);
}

__device__ __forceinline__ v16h cat8(v8h lo, v8h hi) {
  return __builtin_shufflevector(lo, hi, 0,1,2,3,4,5,6,7,8,9,10,11,12,13,14,15);
}

__device__ __forceinline__ float xmax16(float v) {
  v = fmaxf(v, __shfl_xor(v, 1, 32));
  v = fmaxf(v, __shfl_xor(v, 2, 32));
  v = fmaxf(v, __shfl_xor(v, 4, 32));
  v = fmaxf(v, __shfl_xor(v, 8, 32));
  return v;
}
__device__ __forceinline__ float xsum16(float v) {
  v += __shfl_xor(v, 1, 32);
  v += __shfl_xor(v, 2, 32);
  v += __shfl_xor(v, 4, 32);
  v += __shfl_xor(v, 8, 32);
  return v;
}

// -------------------- f32 -> f16 TRANSPOSED weight conversion --------------------
// dst[n*ND + k] = (f16) src[k*ND + n]; B fragments then load contiguous k-runs.
__global__ __launch_bounds__(256)
void k_cvt_transpose(const float* __restrict__ src, _Float16* __restrict__ dst) {
  int i = blockIdx.x * 256 + threadIdx.x;   // 0..65535
  int k = i >> 8, n = i & (ND - 1);
  dst[(size_t)n * ND + k] = (_Float16)src[i];
}

// Load a WMMA B-fragment from a k-contiguous (transposed) f16 matrix.
// row stride = ND halves; lane picks column n, halves are k = kbase + hi*8 + e (e<8), kbase + 16 + hi*8 + (e-8) (e>=8).
__device__ __forceinline__ v16h load_bfragT(const _Float16* __restrict__ WT,
                                            int n, int kbase, int hi) {
  const _Float16* wp = WT + (size_t)n * ND + kbase + hi * 8;
  return cat8(((const v8h*)wp)[0], ((const v8h*)wp)[2]);
}

// -------------------- projection GEMM: X(f32,8192x256) @ W(256x256) --------------------
// Per wave: 16(M) x 64(N) output tile via 4 f32 accumulators; K loop of 8 x 32.
// Store head-split f16: Out[((b*NH+h)*NS + s)*NDK + dk]
__global__ __launch_bounds__(256)
void k_proj_headsplit(const float* __restrict__ X, const _Float16* __restrict__ WT,
                      const float* __restrict__ bias, _Float16* __restrict__ Out) {
  const int lane = threadIdx.x & 31;
  const int wave = threadIdx.x >> 5;
  const int l = lane & 15, hi = lane >> 4;
  const int tile = blockIdx.x * 8 + wave;      // 0..2047
  const int row0 = (tile >> 2) * 16;           // M tile
  const int n0   = (tile & 3) * 64;            // N tile

  v8f acc[4] = { {}, {}, {}, {} };
  const float* ap = X + (size_t)(row0 + l) * ND;
  #pragma unroll
  for (int k0 = 0; k0 < ND; k0 += 32) {
    float4 f0 = *(const float4*)(ap + k0 + hi * 8);
    float4 f1 = *(const float4*)(ap + k0 + hi * 8 + 4);
    float4 f2 = *(const float4*)(ap + k0 + 16 + hi * 8);
    float4 f3 = *(const float4*)(ap + k0 + 16 + hi * 8 + 4);
    v16h a;
    a[0]=(_Float16)f0.x; a[1]=(_Float16)f0.y; a[2]=(_Float16)f0.z; a[3]=(_Float16)f0.w;
    a[4]=(_Float16)f1.x; a[5]=(_Float16)f1.y; a[6]=(_Float16)f1.z; a[7]=(_Float16)f1.w;
    a[8]=(_Float16)f2.x; a[9]=(_Float16)f2.y; a[10]=(_Float16)f2.z; a[11]=(_Float16)f2.w;
    a[12]=(_Float16)f3.x; a[13]=(_Float16)f3.y; a[14]=(_Float16)f3.z; a[15]=(_Float16)f3.w;
    #pragma unroll
    for (int t = 0; t < 4; ++t) {
      v16h bf = load_bfragT(WT, n0 + t * 16 + l, k0, hi);
      acc[t] = wmma_f16(a, bf, acc[t]);
    }
  }
  const int b  = row0 >> 11;
  const int s0 = row0 & (NS - 1);
  #pragma unroll
  for (int t = 0; t < 4; ++t) {
    const int n = n0 + t * 16 + l;
    const int h = n >> 5, dk = n & 31;
    const float bv = bias[n];
    _Float16* op = Out + ((size_t)(b * NH + h) * NS) * NDK + dk;
    #pragma unroll
    for (int r = 0; r < 8; ++r)
      op[(size_t)(s0 + r + 8 * hi) * NDK] = (_Float16)(acc[t][r] + bv);
  }
}

// Same GEMM, but stores V transposed: Vt[((b*NH+h)*NDK + dk)*NS + s]
__global__ __launch_bounds__(256)
void k_proj_vT(const float* __restrict__ X, const _Float16* __restrict__ WT,
               const float* __restrict__ bias, _Float16* __restrict__ Vt) {
  const int lane = threadIdx.x & 31;
  const int wave = threadIdx.x >> 5;
  const int l = lane & 15, hi = lane >> 4;
  const int tile = blockIdx.x * 8 + wave;
  const int row0 = (tile >> 2) * 16;
  const int n0   = (tile & 3) * 64;

  v8f acc[4] = { {}, {}, {}, {} };
  const float* ap = X + (size_t)(row0 + l) * ND;
  #pragma unroll
  for (int k0 = 0; k0 < ND; k0 += 32) {
    float4 f0 = *(const float4*)(ap + k0 + hi * 8);
    float4 f1 = *(const float4*)(ap + k0 + hi * 8 + 4);
    float4 f2 = *(const float4*)(ap + k0 + 16 + hi * 8);
    float4 f3 = *(const float4*)(ap + k0 + 16 + hi * 8 + 4);
    v16h a;
    a[0]=(_Float16)f0.x; a[1]=(_Float16)f0.y; a[2]=(_Float16)f0.z; a[3]=(_Float16)f0.w;
    a[4]=(_Float16)f1.x; a[5]=(_Float16)f1.y; a[6]=(_Float16)f1.z; a[7]=(_Float16)f1.w;
    a[8]=(_Float16)f2.x; a[9]=(_Float16)f2.y; a[10]=(_Float16)f2.z; a[11]=(_Float16)f2.w;
    a[12]=(_Float16)f3.x; a[13]=(_Float16)f3.y; a[14]=(_Float16)f3.z; a[15]=(_Float16)f3.w;
    #pragma unroll
    for (int t = 0; t < 4; ++t) {
      v16h bf = load_bfragT(WT, n0 + t * 16 + l, k0, hi);
      acc[t] = wmma_f16(a, bf, acc[t]);
    }
  }
  const int b  = row0 >> 11;
  const int s0 = row0 & (NS - 1);
  #pragma unroll
  for (int t = 0; t < 4; ++t) {
    const int n = n0 + t * 16 + l;
    const int h = n >> 5, dk = n & 31;
    const float bv = bias[n];
    _Float16* op = Vt + ((size_t)(b * NH + h) * NDK + dk) * NS;
    #pragma unroll
    for (int r = 0; r < 8; ++r)
      op[s0 + r + 8 * hi] = (_Float16)(acc[t][r] + bv);
  }
}

// -------------------- per-row squared norms over DK=32 --------------------
__global__ __launch_bounds__(256)
void k_rownorm(const _Float16* __restrict__ Xh, float* __restrict__ out) {
  int i = blockIdx.x * 256 + threadIdx.x;      // row in [0, NBH*NS)
  const v8h* p = (const v8h*)(Xh + (size_t)i * NDK);
  float s = 0.f;
  #pragma unroll
  for (int j = 0; j < 4; ++j) {
    v8h v = p[j];
    #pragma unroll
    for (int e = 0; e < 8; ++e) { float f = (float)v[e]; s += f * f; }
  }
  out[i] = s;
}

// -------------------- flash attention with L2 scores + clamped softmax --------------------
// scores = (||q||^2 + ||k||^2 - 2 q.k) / sqrt(32); den = max(sum e, exp(-m))
__global__ __launch_bounds__(128)
void k_attn(const _Float16* __restrict__ Qh, const _Float16* __restrict__ Kh,
            const _Float16* __restrict__ Vt, const float* __restrict__ q2g,
            const float* __restrict__ k2g, _Float16* __restrict__ Aout) {
  __shared__ __align__(16) _Float16 ldsP[4][16 * 32];
  const int lane = threadIdx.x & 31;
  const int wave = threadIdx.x >> 5;
  const int l = lane & 15, hi = lane >> 4;
  const int bh = blockIdx.x;                      // 0..31
  const int row0 = (blockIdx.y * 4 + wave) * 16;  // 16 query rows per wave
  const _Float16* Qb = Qh + (size_t)bh * NS * NDK;
  const _Float16* Kb = Kh + (size_t)bh * NS * NDK;
  const _Float16* Vb = Vt + (size_t)bh * NDK * NS;
  const float* q2 = q2g + (size_t)bh * NS;
  const float* k2 = k2g + (size_t)bh * NS;

  v16h qa;
  {
    const _Float16* qp = Qb + (size_t)(row0 + l) * NDK;
    qa = cat8(*(const v8h*)(qp + hi * 8), *(const v8h*)(qp + 16 + hi * 8));
  }
  float q2r[8], mrow[8], drow[8];
  #pragma unroll
  for (int r = 0; r < 8; ++r) {
    q2r[r] = q2[row0 + r + 8 * hi];
    mrow[r] = -__builtin_huge_valf();
    drow[r] = 0.f;
  }
  v8f o0 = {}, o1 = {};
  const float isd = 0.17677669529663687f;  // 1/sqrt(32)
  _Float16* P = ldsP[wave];

  for (int c = 0; c < NS; c += 32) {
    // --- scores tile: two 16x16 WMMAs over a 32-key chunk ---
    const _Float16* kp0 = Kb + (size_t)(c + l) * NDK + hi * 8;
    const _Float16* kp1 = Kb + (size_t)(c + 16 + l) * NDK + hi * 8;
    v16h kb0 = cat8(((const v8h*)kp0)[0], ((const v8h*)kp0)[2]);
    v16h kb1 = cat8(((const v8h*)kp1)[0], ((const v8h*)kp1)[2]);
    v8f z = {};
    v8f s0 = wmma_f16(qa, kb0, z);
    v8f s1 = wmma_f16(qa, kb1, z);
    float k2a = k2[c + l];
    float k2b = k2[c + 16 + l];
    if (c + 32 < NS) {  // global_prefetch_b8 for next chunk
      __builtin_prefetch(Kb + (size_t)(c + 32 + lane) * NDK, 0, 0);
      __builtin_prefetch(Vb + (size_t)lane * NS + c + 32, 0, 0);
    }
    // --- online softmax over 32 columns (lane groups of 16 per row) ---
    #pragma unroll
    for (int r = 0; r < 8; ++r) {
      float sc0 = (q2r[r] + k2a - 2.f * s0[r]) * isd;
      float sc1 = (q2r[r] + k2b - 2.f * s1[r]) * isd;
      float cm  = xmax16(fmaxf(sc0, sc1));
      float mn  = fmaxf(mrow[r], cm);
      float scl = __expf(mrow[r] - mn);
      float p0  = __expf(sc0 - mn);
      float p1  = __expf(sc1 - mn);
      float rs  = xsum16(p0 + p1);
      drow[r] = drow[r] * scl + rs;
      mrow[r] = mn;
      o0[r] *= scl;
      o1[r] *= scl;
      P[(r + 8 * hi) * 32 + l]      = (_Float16)p0;
      P[(r + 8 * hi) * 32 + 16 + l] = (_Float16)p1;
    }
    // C-layout -> A-layout transpose of P via per-wave LDS tile
    asm volatile("s_wait_dscnt 0x0" ::: "memory");
    __builtin_amdgcn_wave_barrier();
    const _Float16* pp = P + l * 32 + hi * 8;
    v16h pa = cat8(*(const v8h*)pp, *(const v8h*)(pp + 16));
    // --- O += P(16x32) * V(32x32): two WMMAs (dk halves) ---
    const _Float16* vp0 = Vb + (size_t)l * NS + c + hi * 8;
    const _Float16* vp1 = Vb + (size_t)(16 + l) * NS + c + hi * 8;
    v16h vb0 = cat8(((const v8h*)vp0)[0], ((const v8h*)vp0)[2]);
    v16h vb1 = cat8(((const v8h*)vp1)[0], ((const v8h*)vp1)[2]);
    o0 = wmma_f16(pa, vb0, o0);
    o1 = wmma_f16(pa, vb1, o1);
    __builtin_amdgcn_wave_barrier();  // keep next-iter LDS stores after these loads
  }
  const int b = bh >> 3, h = bh & 7;
  #pragma unroll
  for (int r = 0; r < 8; ++r) {
    const int srow = row0 + r + 8 * hi;
    float den = fmaxf(drow[r], __expf(-mrow[r]));
    float inv = 1.f / den;
    _Float16* op = Aout + ((size_t)b * NS + srow) * ND + h * NDK;
    op[l]      = (_Float16)(o0[r] * inv);
    op[16 + l] = (_Float16)(o1[r] * inv);
  }
}

// -------------------- output projection: attn(f16) @ Wo(f16,T) + bo -> f32 --------------------
__global__ __launch_bounds__(256)
void k_proj_out(const _Float16* __restrict__ Ah, const _Float16* __restrict__ WT,
                const float* __restrict__ bias, float* __restrict__ Out) {
  const int lane = threadIdx.x & 31;
  const int wave = threadIdx.x >> 5;
  const int l = lane & 15, hi = lane >> 4;
  const int tile = blockIdx.x * 8 + wave;
  const int row0 = (tile >> 2) * 16;
  const int n0   = (tile & 3) * 64;

  v8f acc[4] = { {}, {}, {}, {} };
  const _Float16* ap = Ah + (size_t)(row0 + l) * ND;
  #pragma unroll
  for (int k0 = 0; k0 < ND; k0 += 32) {
    v16h a = cat8(*(const v8h*)(ap + k0 + hi * 8),
                  *(const v8h*)(ap + k0 + 16 + hi * 8));
    #pragma unroll
    for (int t = 0; t < 4; ++t) {
      v16h bf = load_bfragT(WT, n0 + t * 16 + l, k0, hi);
      acc[t] = wmma_f16(a, bf, acc[t]);
    }
  }
  #pragma unroll
  for (int t = 0; t < 4; ++t) {
    const int n = n0 + t * 16 + l;
    const float bv = bias[n];
    #pragma unroll
    for (int r = 0; r < 8; ++r)
      Out[(size_t)(row0 + r + 8 * hi) * ND + n] = acc[t][r] + bv;
  }
}

extern "C" void kernel_launch(void* const* d_in, const int* in_sizes, int n_in,
                              void* d_out, int out_size, void* d_ws, size_t ws_size,
                              hipStream_t stream) {
  (void)in_sizes; (void)n_in; (void)out_size; (void)ws_size;
  const float* queries = (const float*)d_in[0];
  const float* keys    = (const float*)d_in[1];
  const float* values  = (const float*)d_in[2];
  const float* Wq = (const float*)d_in[3];
  const float* bq = (const float*)d_in[4];
  const float* Wk = (const float*)d_in[5];
  const float* bk = (const float*)d_in[6];
  const float* Wv = (const float*)d_in[7];
  const float* bv = (const float*)d_in[8];
  const float* Wo = (const float*)d_in[9];
  const float* bo = (const float*)d_in[10];
  float* out = (float*)d_out;

  char* ws = (char*)d_ws;
  size_t off = 0;
  auto alloc = [&](size_t bytes) -> void* {
    void* p = ws + off;
    off += (bytes + 255) & ~(size_t)255;
    return p;
  };
  const size_t WN = (size_t)ND * ND;               // 65536
  _Float16* WqT = (_Float16*)alloc(WN * 2);
  _Float16* WkT = (_Float16*)alloc(WN * 2);
  _Float16* WvT = (_Float16*)alloc(WN * 2);
  _Float16* WoT = (_Float16*)alloc(WN * 2);
  _Float16* Qh  = (_Float16*)alloc((size_t)NBH * NS * NDK * 2);
  _Float16* Kh  = (_Float16*)alloc((size_t)NBH * NS * NDK * 2);
  _Float16* VtT = (_Float16*)alloc((size_t)NBH * NDK * NS * 2);
  float*    q2  = (float*)alloc((size_t)NBH * NS * 4);
  float*    k2  = (float*)alloc((size_t)NBH * NS * 4);
  _Float16* Att = (_Float16*)alloc((size_t)NROWS * ND * 2);

  k_cvt_transpose<<<256, 256, 0, stream>>>(Wq, WqT);
  k_cvt_transpose<<<256, 256, 0, stream>>>(Wk, WkT);
  k_cvt_transpose<<<256, 256, 0, stream>>>(Wv, WvT);
  k_cvt_transpose<<<256, 256, 0, stream>>>(Wo, WoT);

  // 2048 wave-tiles (512 M-tiles x 4 N-tiles), 8 waves per 256-thread block
  k_proj_headsplit<<<256, 256, 0, stream>>>(queries, WqT, bq, Qh);
  k_proj_headsplit<<<256, 256, 0, stream>>>(keys,    WkT, bk, Kh);
  k_proj_vT       <<<256, 256, 0, stream>>>(values,  WvT, bv, VtT);

  k_rownorm<<<256, 256, 0, stream>>>(Qh, q2);   // NBH*NS = 65536 rows
  k_rownorm<<<256, 256, 0, stream>>>(Kh, k2);

  // 32 (b,h) x 32 blocks x 4 waves x 16 rows = 2048 rows per head
  k_attn<<<dim3(32, 32), 128, 0, stream>>>(Qh, Kh, VtT, q2, k2, Att);

  k_proj_out<<<256, 256, 0, stream>>>(Att, WoT, bo, out);
}
